// SparseResBlock_63496796504736
// MI455X (gfx1250) — compile-verified, matrix operand order checked by reading, not run on hardware
//
#include <hip/hip_runtime.h>
#include <hip/hip_bf16.h>

typedef __attribute__((ext_vector_type(16))) _Float16 v16h;
typedef __attribute__((ext_vector_type(8)))  float    v8f;
typedef __attribute__((ext_vector_type(4)))  unsigned int u32x4;

#define CIN  64
#define COUT 128
#define NK   27
#define EPSV 1e-5f

// ---------------------------------------------------------------------------
// f32 -> f16 convert
__global__ void cvt_f16_kernel(const float* __restrict__ x, _Float16* __restrict__ y, long tot) {
    long e = (long)blockIdx.x * blockDim.x + threadIdx.x;
    if (e < tot) y[e] = (_Float16)x[e];
}

// Zero the BN stats workspace (3 x 256 floats)
__global__ void zero_stats_kernel(float* __restrict__ st) {
    int t = threadIdx.x;
    for (int i = t; i < 768; i += 256) st[i] = 0.0f;
}

// ---------------------------------------------------------------------------
// Repack W [K][Cin][128] f32 into WMMA B-fragments, f16.
// Fragment linear layout: ((((k*KC + kc)*8 + ct)*32 + lane)*16 + j)
// lane<16: col = lane,     j<8 -> K = kc*32 + j,      j>=8 -> K = kc*32 + 16 + (j-8)
// lane>=16: col = lane-16, j<8 -> K = kc*32 + 8 + j,  j>=8 -> K = kc*32 + 24 + (j-8)
__global__ void pack_w_kernel(const float* __restrict__ W, _Float16* __restrict__ dst,
                              int KC, long total) {
    long t = (long)blockIdx.x * blockDim.x + threadIdx.x;
    if (t >= total) return;
    int j    = (int)(t & 15);
    int lane = (int)((t >> 4) & 31);
    int ct   = (int)((t >> 9) & 7);
    int rest = (int)(t >> 12);
    int kc   = rest % KC;
    int k    = rest / KC;
    int ncol = lane & 15;
    int hi   = lane >> 4;
    int kk   = (j < 8) ? (hi * 8 + j) : (16 + hi * 8 + (j - 8));
    int Kidx = kc * 32 + kk;
    int col  = ct * 16 + ncol;
    int Cin  = KC * 32;
    dst[t] = (_Float16)W[((size_t)k * Cin + Kidx) * COUT + col];
}

// ---------------------------------------------------------------------------
// Rulebook sparse conv via WMMA. One wave computes a 32x128 output tile
// (two 16-row groups sharing each B fragment -> 2x register reuse of weights).
// X  : [n][KC*32] f16 row-major features
// Wf : packed B-fragments [nk][KC][8][32][16] f16
// nbr: [nk][n] int32 (-1 = missing neighbor)
// Y  : [n][128] f32 output
template <int KC>
__global__ void spconv_wmma_kernel(const _Float16* __restrict__ X,
                                   const _Float16* __restrict__ Wf,
                                   const int* __restrict__ nbr,
                                   float* __restrict__ Y,
                                   int n, int ntiles, int nk) {
    const int wave = threadIdx.x >> 5;
    const int lane = threadIdx.x & 31;
    const int tile = blockIdx.x * 8 + wave;
    if (tile >= ntiles) return;                 // wave-uniform: EXEC stays all-ones
    const int tb  = tile * 32;
    const int m   = lane & 15;
    const int hi8 = (lane >> 4) << 3;

    v8f acc0[8], acc1[8];
    #pragma unroll
    for (int i = 0; i < 8; i++) {
        acc0[i] = (v8f){0.f,0.f,0.f,0.f,0.f,0.f,0.f,0.f};
        acc1[i] = (v8f){0.f,0.f,0.f,0.f,0.f,0.f,0.f,0.f};
    }

    int rowg0 = tb + m;
    int rowg1 = tb + 16 + m;
    const bool rowok0 = rowg0 < n;
    const bool rowok1 = rowg1 < n;
    if (rowg0 > n - 1) rowg0 = n - 1;
    if (rowg1 > n - 1) rowg1 = n - 1;

    for (int k = 0; k < nk; k++) {
        const int idx0 = nbr[(size_t)k * n + rowg0];
        const int idx1 = nbr[(size_t)k * n + rowg1];
        const bool v0 = rowok0 && (idx0 >= 0);
        const bool v1 = rowok1 && (idx1 >= 0);
        const _Float16 s0 = v0 ? (_Float16)1.0f : (_Float16)0.0f;
        const _Float16 s1 = v1 ? (_Float16)1.0f : (_Float16)0.0f;
        const _Float16* rp0 = X + (size_t)(idx0 < 0 ? 0 : idx0) * (KC * 32);
        const _Float16* rp1 = X + (size_t)(idx1 < 0 ? 0 : idx1) * (KC * 32);
        const _Float16* bpk = Wf + (size_t)(k * KC) * 4096 + (size_t)lane * 16;
        if (k + 1 < nk)  // warm next offset's weight fragments (global_prefetch_b8)
            __builtin_prefetch(Wf + (size_t)((k + 1) * KC) * 4096 + (size_t)lane * 16, 0, 0);

        #pragma unroll
        for (int kc = 0; kc < KC; kc++) {
            union { v16h h; u32x4 q[2]; } a0, a1;
            a0.q[0] = *(const u32x4*)(rp0 + kc * 32 + hi8);
            a0.q[1] = *(const u32x4*)(rp0 + kc * 32 + 16 + hi8);
            a1.q[0] = *(const u32x4*)(rp1 + kc * 32 + hi8);
            a1.q[1] = *(const u32x4*)(rp1 + kc * 32 + 16 + hi8);
            a0.h *= s0;                                   // mask missing neighbors
            a1.h *= s1;
            const _Float16* bp = bpk + (size_t)kc * 4096;
            #pragma unroll
            for (int ct = 0; ct < 8; ct++) {
                union { v16h h; u32x4 q[2]; } b;
                b.q[0] = *(const u32x4*)(bp + ct * 512);
                b.q[1] = *(const u32x4*)(bp + ct * 512 + 8);
                acc0[ct] = __builtin_amdgcn_wmma_f32_16x16x32_f16(
                    false, a0.h, false, b.h, (short)0, acc0[ct], false, false);
                acc1[ct] = __builtin_amdgcn_wmma_f32_16x16x32_f16(
                    false, a1.h, false, b.h, (short)0, acc1[ct], false, false);
            }
        }
    }

    // C layout: VGPR r, lanes 0-15 -> (M = r, N = lane), lanes 16-31 -> (M = 8+r)
    const int col0  = lane & 15;
    const int rbase = (lane >> 4) << 3;
    #pragma unroll
    for (int ct = 0; ct < 8; ct++) {
        union { v8f v; float f[8]; } u0, u1;
        u0.v = acc0[ct];
        u1.v = acc1[ct];
        #pragma unroll
        for (int r = 0; r < 8; r++) {
            const int row0 = tb + rbase + r;
            const int row1 = tb + 16 + rbase + r;
            if (row0 < n) Y[(size_t)row0 * COUT + ct * 16 + col0] = u0.f[r];
            if (row1 < n) Y[(size_t)row1 * COUT + ct * 16 + col0] = u1.f[r];
        }
    }
}

// ---------------------------------------------------------------------------
// Per-channel sum / sumsq  (st[0..127] = sum, st[128..255] = sumsq)
__global__ void stats_kernel(const float* __restrict__ x, float* __restrict__ st, int n) {
    const int c  = threadIdx.x & 127;
    const int rg = threadIdx.x >> 7;
    float s = 0.f, s2 = 0.f;
    const long stride = (long)gridDim.x * 2;
    for (long row = (long)blockIdx.x * 2 + rg; row < n; row += stride) {
        float v = x[row * COUT + c];
        s += v; s2 += v * v;
    }
    atomicAdd(&st[c], s);
    atomicAdd(&st[128 + c], s2);
}

// BN + ReLU, output f16 (feeds conv2)
__global__ void bnrelu_h16_kernel(const float* __restrict__ x, const float* __restrict__ st,
                                  const float* __restrict__ gamma, const float* __restrict__ beta,
                                  _Float16* __restrict__ y, int n) {
    long e = (long)blockIdx.x * blockDim.x + threadIdx.x;
    long tot = (long)n * COUT;
    if (e >= tot) return;
    int c = (int)(e & 127);
    float inv  = 1.0f / (float)n;
    float mean = st[c] * inv;
    float var  = st[128 + c] * inv - mean * mean;
    float v = (x[e] - mean) * rsqrtf(var + EPSV) * gamma[c] + beta[c];
    y[e] = (_Float16)fmaxf(v, 0.0f);
}

// out = relu( BN2(h2) + BNd(skip) )
__global__ void final_kernel(const float* __restrict__ h2, const float* __restrict__ sk,
                             const float* __restrict__ st2, const float* __restrict__ std_,
                             const float* __restrict__ g2, const float* __restrict__ b2,
                             const float* __restrict__ gd, const float* __restrict__ bd,
                             float* __restrict__ out, int n) {
    long e = (long)blockIdx.x * blockDim.x + threadIdx.x;
    long tot = (long)n * COUT;
    if (e >= tot) return;
    int c = (int)(e & 127);
    float inv = 1.0f / (float)n;
    float m2 = st2[c] * inv,  v2 = st2[128 + c] * inv - m2 * m2;
    float md = std_[c] * inv, vd = std_[128 + c] * inv - md * md;
    float a = (h2[e] - m2) * rsqrtf(v2 + EPSV) * g2[c] + b2[c];
    float d = (sk[e] - md) * rsqrtf(vd + EPSV) * gd[c] + bd[c];
    out[e] = fmaxf(a + d, 0.0f);
}

// ---------------------------------------------------------------------------
static inline size_t align256(size_t x) { return (x + 255) & ~(size_t)255; }

extern "C" void kernel_launch(void* const* d_in, const int* in_sizes, int n_in,
                              void* d_out, int out_size, void* d_ws, size_t ws_size,
                              hipStream_t stream) {
    const float* feats = (const float*)d_in[0];
    const int*   nbr1  = (const int*)  d_in[1];
    const int*   nbr2  = (const int*)  d_in[2];
    const float* W1    = (const float*)d_in[3];
    const float* W2    = (const float*)d_in[4];
    const float* Wd    = (const float*)d_in[5];
    const float* g1 = (const float*)d_in[6],  *b1 = (const float*)d_in[7];
    const float* g2 = (const float*)d_in[8],  *b2 = (const float*)d_in[9];
    const float* gd = (const float*)d_in[10], *bd = (const float*)d_in[11];
    float* out = (float*)d_out;

    const int n = in_sizes[0] / CIN;

    // workspace carve-out
    char* ws = (char*)d_ws;
    size_t o = 0;
    _Float16* featsh = (_Float16*)(ws + o); o = align256(o + (size_t)n * CIN * 2);
    _Float16* w1f    = (_Float16*)(ws + o); o = align256(o + (size_t)NK * CIN * COUT * 2);
    _Float16* w2f    = (_Float16*)(ws + o); o = align256(o + (size_t)NK * COUT * COUT * 2);
    _Float16* wdf    = (_Float16*)(ws + o); o = align256(o + (size_t)CIN * COUT * 2);
    float*    h1     = (float*)   (ws + o); o = align256(o + (size_t)n * COUT * 4); // reused for h2
    _Float16* h1h    = (_Float16*)(ws + o); o = align256(o + (size_t)n * COUT * 2);
    float*    hskip  = (float*)   (ws + o); o = align256(o + (size_t)n * COUT * 4);
    float*    stats  = (float*)   (ws + o); o = align256(o + 768 * 4);
    float* st1 = stats, *st2 = stats + 256, *std_ = stats + 512;
    (void)ws_size; (void)n_in; (void)out_size;

    const int ntiles = (n + 31) / 32;
    const int gblk   = (ntiles + 7) / 8;

    zero_stats_kernel<<<1, 256, 0, stream>>>(stats);

    { long tot = (long)n * CIN;
      cvt_f16_kernel<<<(int)((tot + 255) / 256), 256, 0, stream>>>(feats, featsh, tot); }

    { long tot = (long)NK * CIN * COUT;
      pack_w_kernel<<<(int)((tot + 255) / 256), 256, 0, stream>>>(W1, w1f, CIN / 32, tot); }
    { long tot = (long)NK * COUT * COUT;
      pack_w_kernel<<<(int)((tot + 255) / 256), 256, 0, stream>>>(W2, w2f, COUT / 32, tot); }
    { long tot = (long)CIN * COUT;
      pack_w_kernel<<<(int)((tot + 255) / 256), 256, 0, stream>>>(Wd, wdf, CIN / 32, tot); }

    // conv1: h1 = sparse_conv(feats, W1, nbr1)
    spconv_wmma_kernel<2><<<gblk, 256, 0, stream>>>(featsh, w1f, nbr1, h1, n, ntiles, NK);
    stats_kernel<<<512, 256, 0, stream>>>(h1, st1, n);
    { long tot = (long)n * COUT;
      bnrelu_h16_kernel<<<(int)((tot + 255) / 256), 256, 0, stream>>>(h1, st1, g1, b1, h1h, n); }

    // skip = feats @ Wd  (identity gather = row 13 of nbr1)
    spconv_wmma_kernel<2><<<gblk, 256, 0, stream>>>(featsh, wdf, nbr1 + (size_t)13 * n, hskip,
                                                    n, ntiles, 1);

    // conv2: h2 = sparse_conv(h, W2, nbr2)  (h2 overwrites h1 buffer)
    spconv_wmma_kernel<4><<<gblk, 256, 0, stream>>>(h1h, w2f, nbr2, h1, n, ntiles, NK);

    stats_kernel<<<512, 256, 0, stream>>>(h1, st2, n);
    stats_kernel<<<512, 256, 0, stream>>>(hskip, std_, n);

    { long tot = (long)n * COUT;
      final_kernel<<<(int)((tot + 255) / 256), 256, 0, stream>>>(h1, hskip, st2, std_,
                                                                 g2, b2, gd, bd, out, n); }
}
